// ContextualMLPDecryptor_6622839570978
// MI455X (gfx1250) — matrix operand order repaired and checked by reading, not compiled, not run on hardware
//
#include <hip/hip_runtime.h>

// ---- types ------------------------------------------------------------
typedef __attribute__((ext_vector_type(16))) __bf16          v16bf;
typedef __attribute__((ext_vector_type(16))) unsigned short  v16u;
typedef __attribute__((ext_vector_type(8)))  unsigned short  v8u;
typedef __attribute__((ext_vector_type(8)))  float           v8f;
typedef __attribute__((ext_vector_type(4)))  float           v4f;

// ---- problem constants ------------------------------------------------
#define VOCAB   38
#define EMB     128
#define CTX     5
#define SEQ     1024
#define BATCH   64
#define KIN     (CTX*EMB)        // 640
#define ROWS    64               // rows of (B*S) per workgroup
#define WAVES   8
#define THREADS (WAVES*32)

// layer shapes
#define K1 KIN
#define N1 512
#define NT1 32
#define KT1 (K1/32)   // 20
#define K2 512
#define N2 256
#define NT2 16
#define KT2 (K2/32)   // 16
#define K3 256
#define N3 128
#define NT3 8
#define KT3 (K3/32)   // 8
#define K4 128
#define N4 VOCAB
#define NT4 3         // 48 cols, zero padded
#define KT4 (K4/32)   // 4

// packed-weight sizes in ushort elements: Ntiles*Ktiles*32 lanes*16 elems
#define PW1_ELEMS (NT1*KT1*32*16)   // 327680
#define PW2_ELEMS (NT2*KT2*32*16)   // 131072
#define PW3_ELEMS (NT3*KT3*32*16)   //  32768
#define PW4_ELEMS (NT4*KT4*32*16)   //   6144

// round-to-nearest-even f32 -> bf16 (bit pattern)
__device__ __forceinline__ unsigned short f2bf(float f) {
    unsigned u = __builtin_bit_cast(unsigned, f);
    unsigned r = u + 0x7FFFu + ((u >> 16) & 1u);
    return (unsigned short)(r >> 16);
}

// ---- one-time weight pack: fp32 W[N][K] -> bf16 B-fragment layout ------
// Fragment layout used here (dense B, 32x16, contiguous k, not ISA table order): lane 0..15 = col N=lane,
// K[kt*32 .. +15]; lane 16..31 = same col, K[kt*32+16 .. +31]; 2 bf16/VGPR.
// Storage: slot = ((nt*Kt + kt)*32 + lane), 16 contiguous ushorts per slot.
__global__ void __launch_bounds__(256)
pack_weights_kernel(const float* __restrict__ W, unsigned short* __restrict__ dst,
                    int K, int Nreal, int Ntiles) {
    const int Kt = K >> 5;
    const int total = Ntiles * Kt * 32;
    const int idx = blockIdx.x * blockDim.x + threadIdx.x;
    if (idx >= total) return;
    const int lane = idx & 31;
    const int tile = idx >> 5;
    const int kt = tile % Kt;
    const int nt = tile / Kt;
    const int n  = nt * 16 + (lane & 15);
    const int kb = kt * 32 + ((lane >> 4) << 4);

    v8u lo = {}, hi = {};
    if (n < Nreal) {
        const float* w = W + (size_t)n * K + kb;
        v4f f0 = *(const v4f*)(w + 0);
        v4f f1 = *(const v4f*)(w + 4);
        v4f f2 = *(const v4f*)(w + 8);
        v4f f3 = *(const v4f*)(w + 12);
#pragma unroll
        for (int j = 0; j < 4; ++j) {
            lo[j]     = f2bf(f0[j]);
            lo[4 + j] = f2bf(f1[j]);
            hi[j]     = f2bf(f2[j]);
            hi[4 + j] = f2bf(f3[j]);
        }
    }
    v8u* d = (v8u*)(dst + ((size_t)idx << 4));
    d[0] = lo;
    d[1] = hi;
}

__device__ __forceinline__ v16bf combine(v8u lo, v8u hi) {
    v16u r;
#pragma unroll
    for (int i = 0; i < 8; ++i) { r[i] = lo[i]; r[8 + i] = hi[i]; }
    return __builtin_bit_cast(v16bf, r);
}

// ---- one MLP layer: out[ROWS][N] = act( in[ROWS][K] @ W^T + bias ) -----
// Fully compile-time shapes -> pointer-induction inner loop with constant
// immediate offsets for every fragment load. MT x NT register blocking:
// each B fragment is reused by MT WMMAs, each A fragment by NT WMMAs.
template <int K, int Nreal, int Ntiles, int MT, int NT, bool RELU, bool TO_GLOBAL>
__device__ __forceinline__ void mlp_layer(const unsigned short* lds_in,
                                          unsigned short* lds_out,
                                          const unsigned short* __restrict__ Wp,
                                          const float* __restrict__ bias,
                                          float* __restrict__ gout, int rowBase) {
    constexpr int Kt      = K / 32;
    constexpr int ntg     = Ntiles / NT;          // N-tile groups
    constexpr int mtg     = (ROWS / 16) / MT;     // M-tile groups
    constexpr int nGroups = mtg * ntg;
    constexpr int USTR    = Kt * 32 * 16;         // ushorts between adjacent nt

    const int lane = threadIdx.x & 31;
    const int wave = threadIdx.x >> 5;

    for (int g = wave; g < nGroups; g += WAVES) {
        const int mt0 = (g / ntg) * MT;
        const int nt0 = (g % ntg) * NT;

        // loop-invariant bases; constant strides inside the K loop
        const unsigned short* wk = Wp + (size_t)nt0 * USTR + ((size_t)lane << 4);
        const unsigned short* ak = lds_in + (mt0 * 16 + (lane & 15)) * K
                                          + ((lane >> 4) << 4);
        v8f acc[MT][NT] = {};
        for (int kt = 0; kt < Kt; ++kt) {
            v16bf a[MT];
#pragma unroll
            for (int m = 0; m < MT; ++m) {
                const unsigned short* pa = ak + m * (16 * K);   // const offset
                a[m] = combine(*(const v8u*)(pa), *(const v8u*)(pa + 8));
            }
#pragma unroll
            for (int u = 0; u < NT; ++u) {
                const v8u* pb = (const v8u*)(wk + u * USTR);    // const offset
                v16bf b = combine(pb[0], pb[1]);
#pragma unroll
                for (int m = 0; m < MT; ++m)
                    acc[m][u] = __builtin_amdgcn_wmma_f32_16x16x32_bf16(
                        false, a[m], false, b, (short)0, acc[m][u], false, false);
            }
            ak += 32;        // +64 B  : next K-chunk in LDS row
            wk += 32 * 16;   // +1024 B: next kt slot group in packed weights
        }

        // epilogue: C/D layout -> all 8 acc elems of a lane share column n
#pragma unroll
        for (int m = 0; m < MT; ++m) {
#pragma unroll
            for (int u = 0; u < NT; ++u) {
                const int n = (nt0 + u) * 16 + (lane & 15);
                float bn;
                if constexpr (Nreal == Ntiles * 16) bn = bias[n];
                else bn = (n < Nreal) ? bias[n] : 0.0f;
#pragma unroll
                for (int i = 0; i < 8; ++i) {
                    const int row = (mt0 + m) * 16 + i + ((lane >> 4) << 3);
                    float v = acc[m][u][i] + bn;
                    if (RELU) v = v > 0.0f ? v : 0.0f;
                    if constexpr (TO_GLOBAL) {
                        if (n < Nreal)
                            gout[(size_t)(rowBase + row) * VOCAB + n] = v;
                    } else {
                        lds_out[row * (Ntiles * 16) + n] = f2bf(v);
                    }
                }
            }
        }
    }
}

__global__ void __launch_bounds__(THREADS)
ctx_mlp_fused_kernel(const int* __restrict__ enc, const float* __restrict__ emb,
                     const unsigned short* __restrict__ Wp1, const float* __restrict__ b1,
                     const unsigned short* __restrict__ Wp2, const float* __restrict__ b2,
                     const unsigned short* __restrict__ Wp3, const float* __restrict__ b3,
                     const unsigned short* __restrict__ Wp4, const float* __restrict__ b4,
                     float* __restrict__ out) {
    extern __shared__ unsigned short lds[];
    unsigned short* buf0 = lds;               // ROWS*640 (x / h2)
    unsigned short* buf1 = lds + ROWS * KIN;  // ROWS*512 (h1 / h3)

    const int rowBase = blockIdx.x * ROWS;

    // warm L2 with packed W1 while gathering (global_prefetch_b8)
    __builtin_prefetch(Wp1 + (size_t)threadIdx.x * 128, 0, 1);

    // ---- phase 0: gather sliding-window context into bf16 LDS ----------
    for (int q = threadIdx.x; q < (ROWS * KIN) / 4; q += THREADS) {
        const int elem = q * 4;
        const int r = elem / KIN;
        const int c = elem - r * KIN;
        const int j = c >> 7;       // context slot 0..4
        const int e = c & (EMB - 1);
        const int gr = rowBase + r;
        const int bb = gr >> 10;    // batch index
        const int s  = gr & (SEQ - 1);
        const int sp = s + j - 2;
        v4f v = {};
        if (sp >= 0 && sp < SEQ) {
            const int tok = enc[(bb << 10) + sp];
            v = *(const v4f*)(emb + (size_t)tok * EMB + e);
        }
        unsigned short* d = buf0 + elem;
        d[0] = f2bf(v[0]); d[1] = f2bf(v[1]);
        d[2] = f2bf(v[2]); d[3] = f2bf(v[3]);
    }
    __syncthreads();

    // ---- fused MLP chain, activations ping-pong in LDS -----------------
    mlp_layer<K1, N1, NT1, 2, 4, true,  false>(buf0, buf1, Wp1, b1, nullptr, rowBase);
    __syncthreads();
    mlp_layer<K2, N2, NT2, 2, 4, true,  false>(buf1, buf0, Wp2, b2, nullptr, rowBase);
    __syncthreads();
    mlp_layer<K3, N3, NT3, 1, 4, true,  false>(buf0, buf1, Wp3, b3, nullptr, rowBase);
    __syncthreads();
    mlp_layer<K4, N4, NT4, 1, 3, false, true >(buf1, nullptr, Wp4, b4, out, rowBase);
}

extern "C" void kernel_launch(void* const* d_in, const int* in_sizes, int n_in,
                              void* d_out, int out_size, void* d_ws, size_t ws_size,
                              hipStream_t stream) {
    const int*   enc = (const int*)  d_in[0];
    const float* emb = (const float*)d_in[1];
    const float* W1  = (const float*)d_in[2];
    const float* b1  = (const float*)d_in[3];
    const float* W2  = (const float*)d_in[4];
    const float* b2  = (const float*)d_in[5];
    const float* W3  = (const float*)d_in[6];
    const float* b3  = (const float*)d_in[7];
    const float* W4  = (const float*)d_in[8];
    const float* b4  = (const float*)d_in[9];
    float* out = (float*)d_out;

    // packed bf16 weights in workspace (~995 KB total)
    unsigned short* pw1 = (unsigned short*)d_ws;
    unsigned short* pw2 = pw1 + PW1_ELEMS;
    unsigned short* pw3 = pw2 + PW2_ELEMS;
    unsigned short* pw4 = pw3 + PW3_ELEMS;

    // one-time (per launch) weight repack: fp32 -> bf16 fragment layout
    {
        int t1 = NT1 * KT1 * 32, t2 = NT2 * KT2 * 32,
            t3 = NT3 * KT3 * 32, t4 = NT4 * KT4 * 32;
        pack_weights_kernel<<<(t1 + 255) / 256, 256, 0, stream>>>(W1, pw1, K1, N1, NT1);
        pack_weights_kernel<<<(t2 + 255) / 256, 256, 0, stream>>>(W2, pw2, K2, N2, NT2);
        pack_weights_kernel<<<(t3 + 255) / 256, 256, 0, stream>>>(W3, pw3, K3, N3, NT3);
        pack_weights_kernel<<<(t4 + 255) / 256, 256, 0, stream>>>(W4, pw4, K4, N4, NT4);
    }

    const int totalRows = BATCH * SEQ;          // 65536
    dim3 grid(totalRows / ROWS);                // 1024 blocks
    dim3 block(THREADS);                        // 8 wave32s
    size_t shmem = (size_t)(ROWS * KIN + ROWS * 512) * sizeof(unsigned short); // 144 KB

    ctx_mlp_fused_kernel<<<grid, block, shmem, stream>>>(
        enc, emb, pw1, b1, pw2, b2, pw3, b3, pw4, b4, out);
}
